// GPT_47158740910265
// MI455X (gfx1250) — matrix operand order repaired and checked by reading, not compiled, not run on hardware
//
#include <hip/hip_runtime.h>
#include <hip/hip_bf16.h>
#include <cmath>

typedef __bf16 bf16;
typedef __attribute__((ext_vector_type(16))) __bf16 v16bf;
typedef __attribute__((ext_vector_type(8)))  float   v8f;
typedef __attribute__((ext_vector_type(4)))  int     v4i;

union FragBF { v16bf v; uint4 q[2]; };

#define WMMA_BF16(a, b, c) \
  __builtin_amdgcn_wmma_f32_16x16x32_bf16(false, (a), false, (b), (short)0, (c), false, false)

// Async memory->LDS staging (GLOBAL_LOAD_ASYNC_TO_LDS_B128, ASYNCcnt) when the
// toolchain exposes it; otherwise fall back to VGPR-bounce b128 copies.
#if defined(__gfx1250__) && \
    __has_builtin(__builtin_amdgcn_global_load_async_to_lds_b128) && \
    __has_builtin(__builtin_amdgcn_s_wait_asynccnt)
#define ASYNC_LDS 1
#else
#define ASYNC_LDS 0
#endif

__device__ __forceinline__ void stage_b128(const bf16* g, bf16* l) {
#if ASYNC_LDS
  __builtin_amdgcn_global_load_async_to_lds_b128(
      (__attribute__((address_space(1))) v4i*)(uintptr_t)g,
      (__attribute__((address_space(3))) v4i*)(uintptr_t)l, 0, 0);
#else
  *(uint4*)l = *(const uint4*)g;
#endif
}

__device__ __forceinline__ void stage_wait() {
#if ASYNC_LDS
  __builtin_amdgcn_s_wait_asynccnt(0);
#endif
}

// ---------------------------------------------------------------------------
// f32 -> bf16 convert (plain)
// ---------------------------------------------------------------------------
__global__ void f32_to_bf16_kernel(const float* __restrict__ in,
                                   bf16* __restrict__ out, size_t n) {
  size_t i = (size_t)blockIdx.x * blockDim.x + threadIdx.x;
  if (i < n) out[i] = (bf16)in[i];
}

// ---------------------------------------------------------------------------
// f32 [K,N] -> bf16 [N,K] transpose-convert (weights; once per launch)
// ---------------------------------------------------------------------------
__global__ void f32_to_bf16_T_kernel(const float* __restrict__ in,
                                     bf16* __restrict__ out, int K, int N) {
  size_t i = (size_t)blockIdx.x * blockDim.x + threadIdx.x;
  if (i < (size_t)K * N) {
    int k = (int)(i / N), n = (int)(i % N);
    out[(size_t)n * K + k] = (bf16)in[i];
  }
}

// ---------------------------------------------------------------------------
// RMSNorm over rows of length C; writes bf16
// ---------------------------------------------------------------------------
__global__ __launch_bounds__(256)
void rmsnorm_kernel(const float* __restrict__ X, bf16* __restrict__ Out, int C) {
  __shared__ float red[8];
  int row = blockIdx.x;
  const float* x = X + (size_t)row * C;
  float ss = 0.f;
  for (int i = threadIdx.x; i < C; i += 256) { float v = x[i]; ss += v * v; }
  for (int off = 16; off > 0; off >>= 1) ss += __shfl_xor(ss, off, 32);
  int w = threadIdx.x >> 5, lane = threadIdx.x & 31;
  if (lane == 0) red[w] = ss;
  __syncthreads();
  if (threadIdx.x == 0) {
    float tot = 0.f;
    #pragma unroll
    for (int i = 0; i < 8; ++i) tot += red[i];
    red[0] = rsqrtf(tot / (float)C + 1.1920929e-07f);
  }
  __syncthreads();
  float r = red[0];
  bf16* o = Out + (size_t)row * C;
  for (int i = threadIdx.x; i < C; i += 256) o[i] = (bf16)(x[i] * r);
}

// ---------------------------------------------------------------------------
// RoPE + per-head RMS: X [rows,128] f32 -> Out bf16. One wave per head-row.
// ---------------------------------------------------------------------------
__global__ __launch_bounds__(256)
void rope_rms_kernel(const float* __restrict__ X, const float* __restrict__ cosb,
                     const float* __restrict__ sinb, bf16* __restrict__ Out,
                     int T, int NH, int total) {
  int gw = (int)((blockIdx.x * blockDim.x + threadIdx.x) >> 5);
  int lane = threadIdx.x & 31;
  if (gw >= total) return;
  int t = (gw / NH) % T;
  const float* row = X + (size_t)gw * 128;
  const float* cs = cosb + (size_t)t * 64;
  const float* sn = sinb + (size_t)t * 64;
  int i0 = lane * 2;
  float t1a = row[i0],      t1b = row[i0 + 1];
  float t2a = row[64 + i0], t2b = row[64 + i0 + 1];
  float ca = cs[i0], cb = cs[i0 + 1], sa = sn[i0], sb = sn[i0 + 1];
  float o0 =  t1a * ca + t2a * sa;
  float o1 =  t1b * cb + t2b * sb;
  float o2 = -t1a * sa + t2a * ca;
  float o3 = -t1b * sb + t2b * cb;
  float ss = o0 * o0 + o1 * o1 + o2 * o2 + o3 * o3;
  for (int off = 16; off > 0; off >>= 1) ss += __shfl_xor(ss, off, 32);
  float r = rsqrtf(ss / 128.f + 1.1920929e-07f);
  bf16* out = Out + (size_t)gw * 128;
  out[i0]          = (bf16)(o0 * r);
  out[i0 + 1]      = (bf16)(o1 * r);
  out[64 + i0]     = (bf16)(o2 * r);
  out[64 + i0 + 1] = (bf16)(o3 * r);
}

// ---------------------------------------------------------------------------
// Router: logits = xf @ Wr (C x 8), top-2, softmax -> gates [N,8] f32
// ---------------------------------------------------------------------------
__global__ __launch_bounds__(256)
void router_kernel(const bf16* __restrict__ Xf, const float* __restrict__ Wr,
                   float* __restrict__ gates, int C) {
  int gw = (int)((blockIdx.x * 256 + threadIdx.x) >> 5);
  int lane = threadIdx.x & 31;
  const bf16* x = Xf + (size_t)gw * C;
  float acc[8] = {0.f, 0.f, 0.f, 0.f, 0.f, 0.f, 0.f, 0.f};
  for (int c = lane; c < C; c += 32) {
    float xv = (float)x[c];
    const float* wrow = Wr + (size_t)c * 8;
    #pragma unroll
    for (int e = 0; e < 8; ++e) acc[e] += xv * wrow[e];
  }
  #pragma unroll
  for (int e = 0; e < 8; ++e)
    for (int off = 16; off > 0; off >>= 1) acc[e] += __shfl_xor(acc[e], off, 32);
  if (lane == 0) {
    int i0 = 0;
    for (int e = 1; e < 8; ++e) if (acc[e] > acc[i0]) i0 = e;
    int i1 = -1;
    for (int e = 0; e < 8; ++e) {
      if (e == i0) continue;
      if (i1 < 0 || acc[e] > acc[i1]) i1 = e;
    }
    float m = acc[i0];
    float e0 = __expf(acc[i0] - m), e1 = __expf(acc[i1] - m);
    float inv = 1.f / (e0 + e1);
    float* g = gates + (size_t)gw * 8;
    #pragma unroll
    for (int e = 0; e < 8; ++e) g[e] = 0.f;
    g[i0] = e0 * inv;
    g[i1] = e1 * inv;
  }
}

// ---------------------------------------------------------------------------
// Tiled bf16 WMMA GEMM with 2x2 register blocking per wave.
//   C[M,N] = A[M,K] @ W[K,N], with W supplied TRANSPOSED as BT[N,K].
// Block tile 64(M) x 128(N), K-step 32. 8 waves = (wm:2) x (wn:4); each wave
// computes a 32x32 sub-tile = 4 accumulators -> 4 WMMAs per 4 fragment loads.
// EPI: 0 = store f32; 1 = store f32 + addend; 2 = relu(acc)^2 -> bf16;
//      3 = atomicAdd; 4 = atomicAdd(rowscale * acc)
// ---------------------------------------------------------------------------
template <int EPI>
__global__ __launch_bounds__(256)
void gemm_bf16_kernel(const bf16* __restrict__ A, const bf16* __restrict__ BT,
                      float* __restrict__ C, bf16* __restrict__ Cbf,
                      const float* __restrict__ addend,
                      const float* __restrict__ rowscale, int rs_stride,
                      int M, int N, int K) {
  __shared__ bf16 lA[64][40];    // [m][k], 80B rows (16B aligned)
  __shared__ bf16 lB[128][40];   // [n][k]

  const int tid  = threadIdx.x;
  const int lane = tid & 31;
  const int w    = tid >> 5;
  const int wm   = w >> 2;  // 0..1
  const int wn   = w & 3;   // 0..3
  const int m0   = blockIdx.y * 64;
  const int n0   = blockIdx.x * 128;

  v8f acc00 = {}, acc01 = {}, acc10 = {}, acc11 = {};
  for (int k0 = 0; k0 < K; k0 += 32) {
    __syncthreads();
    {   // A tile: 64x32, 2048 elems, 1 x b128 per thread
      int r = tid >> 2;
      int c = (tid & 3) * 8;
      stage_b128(A + (size_t)(m0 + r) * K + k0 + c, &lA[r][c]);
    }
    {   // B tile: 128(n) x 32(k) from BT[N,K], 4096 elems, 2 x b128 per thread
      #pragma unroll
      for (int i = 0; i < 2; ++i) {
        int cid = tid + i * 256;
        int r = cid >> 2;
        int c = (cid & 3) * 8;
        stage_b128(BT + (size_t)(n0 + r) * K + k0 + c, &lB[r][c]);
      }
    }
    stage_wait();
    __syncthreads();

    FragBF fa0, fa1, fb0, fb1;
    {
      int m  = wm * 32 + (lane & 15);
      int ka = (lane < 16) ? 0 : 8;
      const bf16* p0 = &lA[m][ka];
      fa0.q[0] = *(const uint4*)p0;       fa0.q[1] = *(const uint4*)(p0 + 16);
      const bf16* p1 = &lA[m + 16][ka];
      fa1.q[0] = *(const uint4*)p1;       fa1.q[1] = *(const uint4*)(p1 + 16);
    }
    {
      int n  = wn * 32 + (lane & 15);
      int kb = (lane < 16) ? 0 : 8;
      const bf16* p0 = &lB[n][kb];
      fb0.q[0] = *(const uint4*)p0;       fb0.q[1] = *(const uint4*)(p0 + 16);
      const bf16* p1 = &lB[n + 16][kb];
      fb1.q[0] = *(const uint4*)p1;       fb1.q[1] = *(const uint4*)(p1 + 16);
    }
    acc00 = WMMA_BF16(fa0.v, fb0.v, acc00);
    acc01 = WMMA_BF16(fa0.v, fb1.v, acc01);
    acc10 = WMMA_BF16(fa1.v, fb0.v, acc10);
    acc11 = WMMA_BF16(fa1.v, fb1.v, acc11);
  }

  // epilogue; C/D layout: lanes 0-15 -> N=lane, M=r ; lanes 16-31 -> M=8+r
  auto epi = [&](const v8f& acc, int ti, int tj) {
    int nc = n0 + wn * 32 + tj * 16 + (lane & 15);
    int mb = m0 + wm * 32 + ti * 16 + ((lane >> 4) ? 8 : 0);
    #pragma unroll
    for (int r = 0; r < 8; ++r) {
      size_t off = (size_t)(mb + r) * N + nc;
      float v = acc[r];
      if (EPI == 0) {
        C[off] = v;
      } else if (EPI == 1) {
        C[off] = v + addend[off];
      } else if (EPI == 2) {
        float t = v > 0.f ? v : 0.f;
        Cbf[off] = (bf16)(t * t);
      } else if (EPI == 3) {
        atomicAdd(&C[off], v);
      } else if (EPI == 4) {
        atomicAdd(&C[off], rowscale[(size_t)(mb + r) * rs_stride] * v);
      }
    }
  };
  epi(acc00, 0, 0);
  epi(acc01, 0, 1);
  epi(acc10, 1, 0);
  epi(acc11, 1, 1);
}

// ---------------------------------------------------------------------------
// Flash attention (causal, GQA H=16/KVH=4, HD=128), bf16 WMMA.
// Block = 256 thr (8 waves) handles one (b, h, 32-query tile).
// ---------------------------------------------------------------------------
__global__ __launch_bounds__(256)
void attn_kernel(const bf16* __restrict__ Q, const bf16* __restrict__ Kk,
                 const bf16* __restrict__ V, bf16* __restrict__ Y, int T) {
  __shared__ bf16  lQ[32][136];
  __shared__ bf16  lK[64][136];
  __shared__ bf16  lVt[128][72];   // V transposed: [d][key]
  __shared__ bf16  lP[32][72];     // softmaxed probs, bf16
  __shared__ float lS[32][65];     // raw scores f32
  __shared__ float mstat[32], lstat[32], cstat[32];

  const int tid  = threadIdx.x;
  const int lane = tid & 31;
  const int w    = tid >> 5;
  const int wq   = w >> 2;   // 0..1 query sub-tile
  const int wk   = w & 3;    // 0..3: key sub-tile (S) / d-chunk (PV)
  const int q0   = blockIdx.x * 32;
  const int h    = blockIdx.y;
  const int b    = blockIdx.z;
  const int kvh  = h >> 2;   // H/KVH = 4
  const float scale = 0.08838834764831845f; // 1/sqrt(128)

  // Load Q tile (32 x 128)
  #pragma unroll
  for (int i = 0; i < 2; ++i) {
    int c = tid + i * 256;
    int r = c >> 4, col = (c & 15) * 8;
    const bf16* src = Q + ((((size_t)b * T + q0 + r) * 16) + h) * 128 + col;
    *(uint4*)&lQ[r][col] = *(const uint4*)src;
  }
  if (tid < 32) { mstat[tid] = -__builtin_inff(); lstat[tid] = 0.f; }

  v8f acc0 = {}, acc1 = {};
  const int nsteps = (q0 + 32 + 63) / 64;
  for (int s = 0; s < nsteps; ++s) {
    const int k0 = s * 64;
    __syncthreads();
    // K tile (64 x 128): async direct-to-LDS; V tile: load + in-register transpose
    #pragma unroll
    for (int i = 0; i < 4; ++i) {
      int c = tid + i * 256;
      int r = c >> 4, col = (c & 15) * 8;
      size_t base = ((((size_t)b * T + k0 + r) * 4) + kvh) * 128 + col;
      stage_b128(Kk + base, &lK[r][col]);
      uint4 dv = *(const uint4*)(V + base);
      const bf16* ds = (const bf16*)&dv;
      #pragma unroll
      for (int j = 0; j < 8; ++j) lVt[col + j][r] = ds[j];
    }
    stage_wait();
    __syncthreads();

    // S = Q @ K^T  (this wave: q rows wq*16.., keys wk*16..)
    {
      v8f sacc = {};
      int m  = wq * 16 + (lane & 15);
      int ka = (lane < 16) ? 0 : 8;
      int n  = wk * 16 + (lane & 15);
      int kb = (lane < 16) ? 0 : 8;
      #pragma unroll
      for (int d0 = 0; d0 < 128; d0 += 32) {
        FragBF fa, fb;
        const bf16* pa = &lQ[m][d0 + ka];
        fa.q[0] = *(const uint4*)pa;
        fa.q[1] = *(const uint4*)(pa + 16);
        const bf16* pb = &lK[n][d0 + kb];
        fb.q[0] = *(const uint4*)pb;
        fb.q[1] = *(const uint4*)(pb + 16);
        sacc = WMMA_BF16(fa.v, fb.v, sacc);
      }
      int hi = lane >> 4;
      int col = wk * 16 + (lane & 15);
      #pragma unroll
      for (int r = 0; r < 8; ++r)
        lS[wq * 16 + hi * 8 + r][col] = sacc[r] * scale;
    }
    __syncthreads();

    // Online softmax over this 64-key slab; threads 0..31 own one q row each
    if (tid < 32) {
      int qg = q0 + tid;
      float m_old = mstat[tid];
      float rmax = -__builtin_inff();
      for (int j = 0; j < 64; ++j) {
        float v = (k0 + j <= qg) ? lS[tid][j] : -__builtin_inff();
        lS[tid][j] = v;
        rmax = fmaxf(rmax, v);
      }
      float m_new = fmaxf(m_old, rmax);
      float corr = __expf(m_old - m_new);   // 0 on first slab (m_old = -inf)
      float lsum = 0.f;
      for (int j = 0; j < 64; ++j) {
        float p = __expf(lS[tid][j] - m_new);
        lP[tid][j] = (bf16)p;
        lsum += p;
      }
      mstat[tid] = m_new;
      lstat[tid] = lstat[tid] * corr + lsum;
      cstat[tid] = corr;
    }
    __syncthreads();

    // Rescale accumulators, then O += P @ V (d chunks wk*32 and wk*32+16)
    {
      int hi = lane >> 4;
      int rbase = wq * 16 + hi * 8;
      #pragma unroll
      for (int r = 0; r < 8; ++r) {
        float cc = cstat[rbase + r];
        acc0[r] *= cc;
        acc1[r] *= cc;
      }
      int m  = wq * 16 + (lane & 15);
      int ka = (lane < 16) ? 0 : 8;
      int kb = (lane < 16) ? 0 : 8;
      int nl = lane & 15;
      int d0 = wk * 32;
      #pragma unroll
      for (int ks = 0; ks < 64; ks += 32) {
        FragBF fa, fb0, fb1;
        const bf16* pa = &lP[m][ks + ka];
        fa.q[0] = *(const uint4*)pa;
        fa.q[1] = *(const uint4*)(pa + 16);
        const bf16* pb0 = &lVt[d0 + nl][ks + kb];
        fb0.q[0] = *(const uint4*)pb0;
        fb0.q[1] = *(const uint4*)(pb0 + 16);
        acc0 = WMMA_BF16(fa.v, fb0.v, acc0);
        const bf16* pb1 = &lVt[d0 + 16 + nl][ks + kb];
        fb1.q[0] = *(const uint4*)pb1;
        fb1.q[1] = *(const uint4*)(pb1 + 16);
        acc1 = WMMA_BF16(fa.v, fb1.v, acc1);
      }
    }
  }

  // Final normalize and store Y [B,T,H,128] bf16
  {
    int hi = lane >> 4;
    int rbase = wq * 16 + hi * 8;
    int nl = lane & 15;
    #pragma unroll
    for (int r = 0; r < 8; ++r) {
      int row = rbase + r;
      float inv = 1.f / lstat[row];
      size_t base = ((((size_t)b * T + q0 + row) * 16) + h) * 128;
      Y[base + wk * 32 + nl]      = (bf16)(acc0[r] * inv);
      Y[base + wk * 32 + 16 + nl] = (bf16)(acc1[r] * inv);
    }
  }
}

// ---------------------------------------------------------------------------
// Host side
// ---------------------------------------------------------------------------
extern "C" void kernel_launch(void* const* d_in, const int* in_sizes, int n_in,
                              void* d_out, int out_size, void* d_ws, size_t ws_size,
                              hipStream_t stream) {
  constexpr int B = 2, T = 1024, C = 2048, Dm = 512, E = 8;
  constexpr int N = B * T;          // 2048 tokens
  constexpr int KV = 512;           // KVH*HD

  const float* x    = (const float*)d_in[0];
  const float* cosb = (const float*)d_in[1];
  const float* sinb = (const float*)d_in[2];
  const float* Wq   = (const float*)d_in[3];
  const float* Wk   = (const float*)d_in[4];
  const float* Wv   = (const float*)d_in[5];
  const float* Wo   = (const float*)d_in[6];
  const float* Wr   = (const float*)d_in[7];
  const float* Ws1  = (const float*)d_in[8];
  const float* Ws2  = (const float*)d_in[9];
  const float* We1  = (const float*)d_in[10];
  const float* We2  = (const float*)d_in[11];
  float* out = (float*)d_out;

  char* ws = (char*)d_ws;
  size_t cur = 0;
  auto alloc = [&](size_t bytes) -> void* {
    void* p = ws + cur;
    cur = (cur + bytes + 255) & ~(size_t)255;
    return p;
  };
  // Weights stored TRANSPOSED ([N,K]) in bf16 for b128-clean GEMM staging
  bf16* wqT  = (bf16*)alloc((size_t)C * C * 2);
  bf16* wkT  = (bf16*)alloc((size_t)C * KV * 2);
  bf16* wvT  = (bf16*)alloc((size_t)C * KV * 2);
  bf16* woT  = (bf16*)alloc((size_t)C * C * 2);
  bf16* ws1T = (bf16*)alloc((size_t)C * Dm * 2);
  bf16* ws2T = (bf16*)alloc((size_t)Dm * C * 2);
  bf16* we1T = (bf16*)alloc((size_t)E * C * Dm * 2);
  bf16* we2T = (bf16*)alloc((size_t)E * Dm * C * 2);
  bf16* xn_bf  = (bf16*)alloc((size_t)N * C * 2);
  float* qf    = (float*)alloc((size_t)N * C * 4);
  float* kf    = (float*)alloc((size_t)N * KV * 4);
  float* vf    = (float*)alloc((size_t)N * KV * 4);
  bf16* q_bf   = (bf16*)alloc((size_t)N * C * 2);
  bf16* k_bf   = (bf16*)alloc((size_t)N * KV * 2);
  bf16* v_bf   = (bf16*)alloc((size_t)N * KV * 2);
  bf16* y_bf   = (bf16*)alloc((size_t)N * C * 2);
  bf16* xf_bf  = (bf16*)alloc((size_t)N * C * 2);
  bf16* h_bf   = (bf16*)alloc((size_t)N * Dm * 2);
  bf16* he_bf  = (bf16*)alloc((size_t)N * Dm * 2);
  float* gates = (float*)alloc((size_t)N * E * 4);
  (void)ws_size; (void)in_sizes; (void)n_in; (void)out_size;

  auto conv = [&](const float* src, bf16* dst, size_t n) {
    f32_to_bf16_kernel<<<dim3((unsigned)((n + 255) / 256)), 256, 0, stream>>>(src, dst, n);
  };
  auto convT = [&](const float* src, bf16* dst, int K, int Nn) {
    size_t n = (size_t)K * Nn;
    f32_to_bf16_T_kernel<<<dim3((unsigned)((n + 255) / 256)), 256, 0, stream>>>(src, dst, K, Nn);
  };
  auto gemm = [&](int epi, const bf16* A, const bf16* BT, float* Cc, bf16* Cb,
                  const float* add, const float* rs, int rss, int M, int Nn, int K) {
    dim3 g((unsigned)(Nn / 128), (unsigned)(M / 64));
    switch (epi) {
      case 0: gemm_bf16_kernel<0><<<g, 256, 0, stream>>>(A, BT, Cc, Cb, add, rs, rss, M, Nn, K); break;
      case 1: gemm_bf16_kernel<1><<<g, 256, 0, stream>>>(A, BT, Cc, Cb, add, rs, rss, M, Nn, K); break;
      case 2: gemm_bf16_kernel<2><<<g, 256, 0, stream>>>(A, BT, Cc, Cb, add, rs, rss, M, Nn, K); break;
      case 3: gemm_bf16_kernel<3><<<g, 256, 0, stream>>>(A, BT, Cc, Cb, add, rs, rss, M, Nn, K); break;
      case 4: gemm_bf16_kernel<4><<<g, 256, 0, stream>>>(A, BT, Cc, Cb, add, rs, rss, M, Nn, K); break;
    }
  };

  // 1) One-time transpose-convert of all weights (amortized across 32 M-tiles)
  convT(Wq,  wqT,  C,  C);
  convT(Wk,  wkT,  C,  KV);
  convT(Wv,  wvT,  C,  KV);
  convT(Wo,  woT,  C,  C);
  convT(Ws1, ws1T, C,  Dm);
  convT(Ws2, ws2T, Dm, C);
  for (int e = 0; e < E; ++e) {
    convT(We1 + (size_t)e * C * Dm, we1T + (size_t)e * C * Dm, C,  Dm);
    convT(We2 + (size_t)e * Dm * C, we2T + (size_t)e * Dm * C, Dm, C);
  }

  // 2) xn = rms(x)
  rmsnorm_kernel<<<N, 256, 0, stream>>>(x, xn_bf, C);

  // 3) QKV projections (WMMA)
  gemm(0, xn_bf, wqT, qf, nullptr, nullptr, nullptr, 0, N, C,  C);
  gemm(0, xn_bf, wkT, kf, nullptr, nullptr, nullptr, 0, N, KV, C);
  gemm(0, xn_bf, wvT, vf, nullptr, nullptr, nullptr, 0, N, KV, C);

  // 4) RoPE + per-head RMS -> bf16 ; V -> bf16
  rope_rms_kernel<<<(N * 16) / 8, 256, 0, stream>>>(qf, cosb, sinb, q_bf, T, 16, N * 16);
  rope_rms_kernel<<<(N * 4)  / 8, 256, 0, stream>>>(kf, cosb, sinb, k_bf, T, 4,  N * 4);
  conv(vf, v_bf, (size_t)N * KV);

  // 5) Flash attention (WMMA, causal, GQA)
  attn_kernel<<<dim3(T / 32, 16, B), 256, 0, stream>>>(q_bf, k_bf, v_bf, y_bf, T);

  // 6) x2 = x + y @ Wo  ->  d_out
  gemm(1, y_bf, woT, out, nullptr, x, nullptr, 0, N, C, C);

  // 7) xf = rms(x2)
  rmsnorm_kernel<<<N, 256, 0, stream>>>(out, xf_bf, C);

  // 8) Router: top-2 gates
  router_kernel<<<N / 8, 256, 0, stream>>>(xf_bf, Wr, gates, C);

  // 9) Shared MLP: d_out += relu(xf@Ws1)^2 @ Ws2
  gemm(2, xf_bf, ws1T, nullptr, h_bf, nullptr, nullptr, 0, N, Dm, C);
  gemm(3, h_bf, ws2T, out, nullptr, nullptr, nullptr, 0, N, C, Dm);

  // 10) Routed experts: d_out += gates[:,e] * (relu(xf@We1[e])^2 @ We2[e])
  for (int e = 0; e < E; ++e) {
    gemm(2, xf_bf, we1T + (size_t)e * C * Dm, nullptr, he_bf,
         nullptr, nullptr, 0, N, Dm, C);
    gemm(4, he_bf, we2T + (size_t)e * Dm * C, out, nullptr,
         nullptr, gates + e, E, N, C, Dm);
  }
}
